// EncoderLayer_90718299226385
// MI455X (gfx1250) — compile-verified, matrix operand order checked by reading, not run on hardware
//
#include <hip/hip_runtime.h>

// ---------------------------------------------------------------------------
// MI455X / gfx1250 encoder layer. All GEMMs run on v_wmma_f32_16x16x32_bf16
// (wave32, 16x16 tiles, fp32 accumulate). GEMM staging uses the gfx1250
// async direct-to-LDS DMA path (GLOBAL_LOAD_ASYNC_TO_LDS_B128 / ASYNCcnt)
// with double-buffered LDS tiles. Attention is flash-style (no SxS score
// tensor in memory).
// ---------------------------------------------------------------------------

typedef __bf16 bf16;
typedef __attribute__((ext_vector_type(16))) __bf16 v16bf;
typedef __attribute__((ext_vector_type(8)))  float  v8f;

#define D_MODEL 1024
#define SEQ     2048
#define BATCH   2
#define NHEAD   16
#define DK      64
#define DFF     4096
#define MTOK    (BATCH * SEQ)   // 4096 rows

__device__ __forceinline__ v8f wmma_bf16(v16bf a, v16bf b, v8f c) {
    return __builtin_amdgcn_wmma_f32_16x16x32_bf16(
        /*neg_a=*/false, a, /*neg_b=*/false, b,
        /*c_mod=*/(short)0, c, /*reuse_a=*/false, /*reuse_b=*/false);
}

// Build a 16x32 bf16 fragment (A or B operand) from two 16-byte chunks.
// Per ISA layout: lanes 0-15 hold K = base+0..7 and base+16..23,
// lanes 16-31 hold K = base+8..15 and base+24..31 -> caller passes
// (ptr + hi*8) and (ptr + 16 + hi*8).
__device__ __forceinline__ v16bf load_frag(const bf16* p0, const bf16* p1) {
    union { v16bf v; uint4 q[2]; } u;
    u.q[0] = *(const uint4*)p0;
    u.q[1] = *(const uint4*)p1;
    return u.v;
}

// gfx1250 async DMA: 16B global -> LDS, tracked by ASYNCcnt (in-order).
// GV mode (SADDR = off). LDS destination address = addr[31:0] of the
// generic pointer (LDS aperture mapping).
__device__ __forceinline__ void async_load_b128(const bf16* gptr, bf16* lptr) {
    unsigned ldsoff = (unsigned)(unsigned long long)lptr;
    asm volatile("global_load_async_to_lds_b128 %0, %1, off"
                 :: "v"(ldsoff), "v"(gptr) : "memory");
}
#define WAIT_ASYNCCNT(n) asm volatile("s_wait_asynccnt %0" :: "i"(n) : "memory")

// ---------------------------------------------------------------------------
// fp32 -> bf16 elementwise convert (n multiple of 4)
// ---------------------------------------------------------------------------
__global__ __launch_bounds__(256) void conv_kernel(const float* __restrict__ X,
                                                   bf16* __restrict__ Y, int n) {
    int idx = (blockIdx.x * 256 + threadIdx.x) * 4;
    if (idx < n) {
        float4 v = *(const float4*)&X[idx];
        Y[idx + 0] = (bf16)v.x;
        Y[idx + 1] = (bf16)v.y;
        Y[idx + 2] = (bf16)v.z;
        Y[idx + 3] = (bf16)v.w;
    }
}

// ---------------------------------------------------------------------------
// fp32 [K,N] -> bf16 [N,K] transpose-convert (32x32 LDS tiles, block 32x8)
// ---------------------------------------------------------------------------
__global__ __launch_bounds__(256) void convT_kernel(const float* __restrict__ W,
                                                    bf16* __restrict__ Wt,
                                                    int K, int N) {
    __shared__ float t[32][33];
    int bn = blockIdx.x * 32, bk = blockIdx.y * 32;
    int tx = threadIdx.x, ty = threadIdx.y;
    for (int i = 0; i < 4; ++i)
        t[ty + i * 8][tx] = W[(size_t)(bk + ty + i * 8) * N + bn + tx];
    __syncthreads();
    for (int i = 0; i < 4; ++i)
        Wt[(size_t)(bn + ty + i * 8) * K + bk + tx] = (bf16)t[tx][ty + i * 8];
}

// ---------------------------------------------------------------------------
// GEMM: C[M,N] = A[M,K](bf16,row) @ Bt[N,K](bf16,row, i.e. B pre-transposed)
//       + bias[N]
// Block: 256 thr (8 waves), tile 128x128x32, double-buffered async-to-LDS.
// Wave grid 2x4 -> 64x32 per wave -> 4x2 accumulators of 16x16.
// Epilogue modes:
//   0: fp32 row-major    1: bf16 row-major    2: bf16 row-major + ReLU
//   3: bf16 per-head transposed [B,H,dk,S]  (for V)
// ---------------------------------------------------------------------------
#define GLDS 40   // padded LDS row stride (halves) -> conflict-free b128 reads

__global__ __launch_bounds__(256)
void gemm_bf16_kernel(const bf16* __restrict__ A, const bf16* __restrict__ Bt,
                      const float* __restrict__ bias,
                      float* __restrict__ Cf, bf16* __restrict__ Cb,
                      int M, int N, int K, int mode) {
    __shared__ bf16 sA[2][128 * GLDS];
    __shared__ bf16 sB[2][128 * GLDS];

    const int tid  = threadIdx.x;
    const int lane = tid & 31, wave = tid >> 5;
    const int wm = wave >> 2, wn = wave & 3;   // 2 x 4 wave grid
    const int l = lane & 15, hi = lane >> 4;
    const int bm = blockIdx.y * 128, bn = blockIdx.x * 128;

    // this thread's two 16B staging chunks (out of 512 per tile side)
    const int row0 = tid >> 2,          off0 = (tid & 3) * 8;
    const int row1 = (tid + 256) >> 2,  off1 = ((tid + 256) & 3) * 8;

    v8f acc[4][2];
    v8f z = {};
    for (int mt = 0; mt < 4; ++mt)
        for (int nt = 0; nt < 2; ++nt) acc[mt][nt] = z;

    const int nk = K >> 5;   // K / 32

    // prologue: fill buffer 0 (4 async ops per thread, per-wave ASYNCcnt += 4)
    {
        async_load_b128(&A[(size_t)(bm + row0) * K + off0],  &sA[0][row0 * GLDS + off0]);
        async_load_b128(&A[(size_t)(bm + row1) * K + off1],  &sA[0][row1 * GLDS + off1]);
        async_load_b128(&Bt[(size_t)(bn + row0) * K + off0], &sB[0][row0 * GLDS + off0]);
        async_load_b128(&Bt[(size_t)(bn + row1) * K + off1], &sB[0][row1 * GLDS + off1]);
    }

    for (int kt = 0; kt < nk; ++kt) {
        const int cur = kt & 1;
        if (kt + 1 < nk) {
            // prefetch next tile into the other buffer (safe: trailing barrier
            // of iteration kt-1 means no wave still reads buffer cur^1)
            const int nxt = cur ^ 1;
            const int k1 = (kt + 1) << 5;
            async_load_b128(&A[(size_t)(bm + row0) * K + k1 + off0],  &sA[nxt][row0 * GLDS + off0]);
            async_load_b128(&A[(size_t)(bm + row1) * K + k1 + off1],  &sA[nxt][row1 * GLDS + off1]);
            async_load_b128(&Bt[(size_t)(bn + row0) * K + k1 + off0], &sB[nxt][row0 * GLDS + off0]);
            async_load_b128(&Bt[(size_t)(bn + row1) * K + k1 + off1], &sB[nxt][row1 * GLDS + off1]);
            // in-order completion: <=4 outstanding -> current tile's 4 done
            WAIT_ASYNCCNT(4);
        } else {
            WAIT_ASYNCCNT(0);
        }
        __syncthreads();   // all waves' fills of buffer `cur` complete

        v16bf af[4], bf[2];
        for (int mt = 0; mt < 4; ++mt) {
            int row = wm * 64 + mt * 16 + l;
            af[mt] = load_frag(&sA[cur][row * GLDS + hi * 8],
                               &sA[cur][row * GLDS + 16 + hi * 8]);
        }
        for (int nt = 0; nt < 2; ++nt) {
            int row = wn * 32 + nt * 16 + l;
            bf[nt] = load_frag(&sB[cur][row * GLDS + hi * 8],
                               &sB[cur][row * GLDS + 16 + hi * 8]);
        }
        for (int mt = 0; mt < 4; ++mt)
            for (int nt = 0; nt < 2; ++nt)
                acc[mt][nt] = wmma_bf16(af[mt], bf[nt], acc[mt][nt]);
        __syncthreads();   // everyone done reading buffer `cur`
    }

    // epilogue (C layout: lanes 0-15 -> M=r, lanes 16-31 -> M=r+8, N=l)
    for (int mt = 0; mt < 4; ++mt)
        for (int nt = 0; nt < 2; ++nt)
            for (int r = 0; r < 8; ++r) {
                int gm = bm + wm * 64 + mt * 16 + r + hi * 8;
                int gn = bn + wn * 32 + nt * 16 + l;
                float v = acc[mt][nt][r] + bias[gn];
                if (mode == 0) {
                    Cf[(size_t)gm * N + gn] = v;
                } else if (mode == 1) {
                    Cb[(size_t)gm * N + gn] = (bf16)v;
                } else if (mode == 2) {
                    Cb[(size_t)gm * N + gn] = (bf16)fmaxf(v, 0.0f);
                } else {  // per-head transpose: [B,H,dk,S]
                    int b = gm >> 11, s = gm & 2047;
                    int h = gn >> 6,  d = gn & 63;
                    Cb[(((size_t)((b * NHEAD + h) * DK + d)) << 11) + s] = (bf16)v;
                }
            }
}

// ---------------------------------------------------------------------------
// Flash attention. Grid: (SEQ/128, BATCH*NHEAD). 8 waves; wave = 16 queries.
// Q,K layout: [B*S, D_MODEL] bf16 (head h at cols h*64..). V layout: [B,H,dk,S].
// Streams 32-key tiles: S=QK^T (4 wmma) -> online softmax -> P via LDS ->
// ctx += P@V (4 wmma). Output ctx: [B*S, D_MODEL] bf16 (merged heads).
// ---------------------------------------------------------------------------
__global__ __launch_bounds__(256)
void attention_kernel(const bf16* __restrict__ Q, const bf16* __restrict__ Km,
                      const bf16* __restrict__ Vt, bf16* __restrict__ ctx) {
    __shared__ bf16 sP[8 * 16 * GLDS];

    const int tid  = threadIdx.x;
    const int lane = tid & 31, wave = tid >> 5;
    const int l = lane & 15, hi = lane >> 4;
    const int bh = blockIdx.y;              // b*NHEAD + h
    const int b  = bh >> 4, h = bh & 15;
    const int q0 = blockIdx.x * 128 + wave * 16;
    const size_t qkBase = ((size_t)b * SEQ) * D_MODEL + h * DK;
    bf16* myP = &sP[wave * 16 * GLDS];

    // Q fragments (16 queries x 64 dk -> two K=32 fragments)
    v16bf aq0, aq1;
    {
        const bf16* qr = Q + qkBase + (size_t)(q0 + l) * D_MODEL;
        aq0 = load_frag(qr + hi * 8,      qr + 16 + hi * 8);
        aq1 = load_frag(qr + 32 + hi * 8, qr + 48 + hi * 8);
    }

    float mrow[8], lrow[8];
    v8f z = {};
    v8f acc[4];
    for (int r = 0; r < 8; ++r) { mrow[r] = -3.0e38f; lrow[r] = 0.0f; }
    for (int g = 0; g < 4; ++g) acc[g] = z;

    for (int kt = 0; kt < SEQ; kt += 32) {
        // ---- scores: 16q x 32k, contraction over dk=64 ----
        v8f c0 = z, c1 = z;
        const bf16* kr0 = Km + qkBase + (size_t)(kt + l) * D_MODEL;
        const bf16* kr1 = Km + qkBase + (size_t)(kt + 16 + l) * D_MODEL;
        v16bf bk00 = load_frag(kr0 + hi * 8,      kr0 + 16 + hi * 8);
        v16bf bk01 = load_frag(kr0 + 32 + hi * 8, kr0 + 48 + hi * 8);
        v16bf bk10 = load_frag(kr1 + hi * 8,      kr1 + 16 + hi * 8);
        v16bf bk11 = load_frag(kr1 + 32 + hi * 8, kr1 + 48 + hi * 8);
        c0 = wmma_bf16(aq0, bk00, c0);
        c0 = wmma_bf16(aq1, bk01, c0);
        c1 = wmma_bf16(aq0, bk10, c1);
        c1 = wmma_bf16(aq1, bk11, c1);

        // ---- online softmax (row = r + hi*8; 16 cols per half-wave) ----
        float p0[8], p1[8], alpha[8];
        for (int r = 0; r < 8; ++r) {
            float s0 = c0[r] * 0.125f;   // 1/sqrt(64)
            float s1 = c1[r] * 0.125f;
            float mx = fmaxf(s0, s1);
            for (int d = 1; d < 16; d <<= 1) mx = fmaxf(mx, __shfl_xor(mx, d, 16));
            float mn = fmaxf(mrow[r], mx);
            alpha[r] = __expf(mrow[r] - mn);
            p0[r] = __expf(s0 - mn);
            p1[r] = __expf(s1 - mn);
            float sum = p0[r] + p1[r];
            for (int d = 1; d < 16; d <<= 1) sum += __shfl_xor(sum, d, 16);
            lrow[r] = lrow[r] * alpha[r] + sum;
            mrow[r] = mn;
        }
        for (int g = 0; g < 4; ++g)
            for (int r = 0; r < 8; ++r) acc[g][r] *= alpha[r];

        // ---- P (16x32) through wave-private LDS into A-fragment layout ----
        for (int r = 0; r < 8; ++r) {
            int row = r + hi * 8;
            myP[row * GLDS + l]      = (bf16)p0[r];
            myP[row * GLDS + 16 + l] = (bf16)p1[r];
        }
        v16bf pa = load_frag(&myP[l * GLDS + hi * 8],
                             &myP[l * GLDS + 16 + hi * 8]);

        // ---- ctx += P @ V  (V pre-transposed: keys contiguous) ----
        for (int g = 0; g < 4; ++g) {
            const bf16* vr = Vt + (((size_t)(bh * DK + g * 16 + l)) << 11) + kt;
            v16bf bv = load_frag(vr + hi * 8, vr + 16 + hi * 8);
            acc[g] = wmma_bf16(pa, bv, acc[g]);
        }
    }

    // ---- normalize + store merged-head ctx ----
    for (int g = 0; g < 4; ++g)
        for (int r = 0; r < 8; ++r) {
            int row = q0 + r + hi * 8;
            int col = h * DK + g * 16 + l;
            ctx[((size_t)(b * SEQ + row)) * D_MODEL + col] =
                (bf16)(acc[g][r] / lrow[r]);
        }
}

// ---------------------------------------------------------------------------
// Residual add + LayerNorm over last dim (1024). One block per row.
// Writes fp32 out; optionally bf16 copy for the next GEMM.
// ---------------------------------------------------------------------------
__global__ __launch_bounds__(256)
void ln_kernel(const float* __restrict__ X, const float* __restrict__ R,
               const float* __restrict__ gamma, const float* __restrict__ beta,
               float* __restrict__ outF, bf16* __restrict__ outB) {
    __shared__ float red[16];
    const int row = blockIdx.x;
    const int tid = threadIdx.x;
    const int lane = tid & 31, wave = tid >> 5;
    const float* xr = X + (size_t)row * D_MODEL;
    const float* rr = R + (size_t)row * D_MODEL;

    float v[4], s = 0.0f, ss = 0.0f;
    for (int i = 0; i < 4; ++i) {
        int c = tid + i * 256;
        v[i] = xr[c] + rr[c];
        s  += v[i];
        ss += v[i] * v[i];
    }
    for (int d = 1; d < 32; d <<= 1) {
        s  += __shfl_xor(s, d, 32);
        ss += __shfl_xor(ss, d, 32);
    }
    if (lane == 0) { red[wave] = s; red[8 + wave] = ss; }
    __syncthreads();
    float ts = 0.0f, tss = 0.0f;
    for (int w = 0; w < 8; ++w) { ts += red[w]; tss += red[8 + w]; }
    float mean = ts * (1.0f / D_MODEL);
    float var  = tss * (1.0f / D_MODEL) - mean * mean;
    float rstd = rsqrtf(var + 1e-5f);
    for (int i = 0; i < 4; ++i) {
        int c = tid + i * 256;
        float y = (v[i] - mean) * rstd * gamma[c] + beta[c];
        outF[(size_t)row * D_MODEL + c] = y;
        if (outB) outB[(size_t)row * D_MODEL + c] = (bf16)y;
    }
}

// ---------------------------------------------------------------------------
// Host orchestration
// ---------------------------------------------------------------------------
extern "C" void kernel_launch(void* const* d_in, const int* in_sizes, int n_in,
                              void* d_out, int out_size, void* d_ws, size_t ws_size,
                              hipStream_t stream) {
    const float* x   = (const float*)d_in[0];
    const float* Wq  = (const float*)d_in[1];  const float* bq  = (const float*)d_in[2];
    const float* Wk  = (const float*)d_in[3];  const float* bk  = (const float*)d_in[4];
    const float* Wv  = (const float*)d_in[5];  const float* bv  = (const float*)d_in[6];
    const float* Wo  = (const float*)d_in[7];  const float* bo  = (const float*)d_in[8];
    const float* W1  = (const float*)d_in[9];  const float* b1  = (const float*)d_in[10];
    const float* W2  = (const float*)d_in[11]; const float* b2  = (const float*)d_in[12];
    const float* g1  = (const float*)d_in[13]; const float* be1 = (const float*)d_in[14];
    const float* g2  = (const float*)d_in[15]; const float* be2 = (const float*)d_in[16];
    float* out = (float*)d_out;

    size_t off = 0;
    char* base = (char*)d_ws;
    auto wsalloc = [&](size_t bytes) -> void* {
        void* p = base + off;
        off = (off + bytes + 255) & ~(size_t)255;
        return p;
    };

    bf16* xb   = (bf16*)wsalloc((size_t)MTOK * D_MODEL * 2);
    bf16* Wqt  = (bf16*)wsalloc((size_t)D_MODEL * D_MODEL * 2);
    bf16* Wkt  = (bf16*)wsalloc((size_t)D_MODEL * D_MODEL * 2);
    bf16* Wvt  = (bf16*)wsalloc((size_t)D_MODEL * D_MODEL * 2);
    bf16* Wot  = (bf16*)wsalloc((size_t)D_MODEL * D_MODEL * 2);
    bf16* W1t  = (bf16*)wsalloc((size_t)DFF * D_MODEL * 2);
    bf16* W2t  = (bf16*)wsalloc((size_t)D_MODEL * DFF * 2);
    bf16* Qb   = (bf16*)wsalloc((size_t)MTOK * D_MODEL * 2);
    bf16* Kb   = (bf16*)wsalloc((size_t)MTOK * D_MODEL * 2);
    bf16* Vtp  = (bf16*)wsalloc((size_t)MTOK * D_MODEL * 2);
    bf16* ctx  = (bf16*)wsalloc((size_t)MTOK * D_MODEL * 2);
    float* aout= (float*)wsalloc((size_t)MTOK * D_MODEL * 4);
    float* hfp = (float*)wsalloc((size_t)MTOK * D_MODEL * 4);
    bf16* hb   = (bf16*)wsalloc((size_t)MTOK * D_MODEL * 2);
    bf16* ffn1 = (bf16*)wsalloc((size_t)MTOK * DFF * 2);
    float* ffn2= (float*)wsalloc((size_t)MTOK * D_MODEL * 4);
    (void)ws_size; (void)in_sizes; (void)n_in; (void)out_size;

    // --- precision conversion (+ weight transpose so GEMM-B is [N,K]) ---
    {
        int n = MTOK * D_MODEL;
        conv_kernel<<<dim3(n / 1024), dim3(256), 0, stream>>>(x, xb, n);
    }
    dim3 tb(32, 8);
    convT_kernel<<<dim3(D_MODEL / 32, D_MODEL / 32), tb, 0, stream>>>(Wq, Wqt, D_MODEL, D_MODEL);
    convT_kernel<<<dim3(D_MODEL / 32, D_MODEL / 32), tb, 0, stream>>>(Wk, Wkt, D_MODEL, D_MODEL);
    convT_kernel<<<dim3(D_MODEL / 32, D_MODEL / 32), tb, 0, stream>>>(Wv, Wvt, D_MODEL, D_MODEL);
    convT_kernel<<<dim3(D_MODEL / 32, D_MODEL / 32), tb, 0, stream>>>(Wo, Wot, D_MODEL, D_MODEL);
    convT_kernel<<<dim3(DFF / 32, D_MODEL / 32),     tb, 0, stream>>>(W1, W1t, D_MODEL, DFF);
    convT_kernel<<<dim3(D_MODEL / 32, DFF / 32),     tb, 0, stream>>>(W2, W2t, DFF, D_MODEL);

    // --- QKV projections (WMMA bf16) ---
    dim3 blk(256);
    gemm_bf16_kernel<<<dim3(D_MODEL / 128, MTOK / 128), blk, 0, stream>>>(
        xb, Wqt, bq, nullptr, Qb, MTOK, D_MODEL, D_MODEL, /*mode=*/1);
    gemm_bf16_kernel<<<dim3(D_MODEL / 128, MTOK / 128), blk, 0, stream>>>(
        xb, Wkt, bk, nullptr, Kb, MTOK, D_MODEL, D_MODEL, /*mode=*/1);
    gemm_bf16_kernel<<<dim3(D_MODEL / 128, MTOK / 128), blk, 0, stream>>>(
        xb, Wvt, bv, nullptr, Vtp, MTOK, D_MODEL, D_MODEL, /*mode=*/3);

    // --- flash attention ---
    attention_kernel<<<dim3(SEQ / 128, BATCH * NHEAD), blk, 0, stream>>>(Qb, Kb, Vtp, ctx);

    // --- output projection (fp32 out) ---
    gemm_bf16_kernel<<<dim3(D_MODEL / 128, MTOK / 128), blk, 0, stream>>>(
        ctx, Wot, bo, aout, nullptr, MTOK, D_MODEL, D_MODEL, /*mode=*/0);

    // --- add & norm 1 (fp32 h, bf16 copy for FFN) ---
    ln_kernel<<<dim3(MTOK), blk, 0, stream>>>(x, aout, g1, be1, hfp, hb);

    // --- FFN ---
    gemm_bf16_kernel<<<dim3(DFF / 128, MTOK / 128), blk, 0, stream>>>(
        hb, W1t, b1, nullptr, ffn1, MTOK, DFF, D_MODEL, /*mode=*/2);  // +ReLU
    gemm_bf16_kernel<<<dim3(D_MODEL / 128, MTOK / 128), blk, 0, stream>>>(
        ffn1, W2t, b2, ffn2, nullptr, MTOK, D_MODEL, DFF, /*mode=*/0);

    // --- add & norm 2 -> final output ---
    ln_kernel<<<dim3(MTOK), blk, 0, stream>>>(hfp, ffn2, g2, be2, out, nullptr);
}
